// MTL_20976620274113
// MI455X (gfx1250) — matrix operand order repaired and checked by reading, not compiled, not run on hardware
//
#include <hip/hip_runtime.h>
#include <hip/hip_bf16.h>

typedef _Float16 h16;
typedef __attribute__((ext_vector_type(16))) _Float16 v16h;
typedef __attribute__((ext_vector_type(8)))  _Float16 v8h;
typedef __attribute__((ext_vector_type(8)))  float    v8f;

#define WMMA16(a,b,c) __builtin_amdgcn_wmma_f32_16x16x32_f16(false,(a),false,(b),(short)0,(c),false,false)

// ---------------- WMMA tile helpers (wave32, 16x16x32 f16->f32) ----------------
// A tile 16x32 (MxK): lane l<16: row=l, K 0..15; lane l>=16: row=l-16,
// K 16..31. p points at this lane's first element for k0=0.
static __device__ __forceinline__ v16h ldA(const h16* p){
  v8h lo = *(const v8h*)p;
  v8h hi = *(const v8h*)(p + 8);
  v16h a;
#pragma unroll
  for (int i = 0; i < 8; ++i){ a[i] = lo[i]; a[8 + i] = hi[i]; }
  return a;
}

// 16x64 output per wave, double-buffered, pointer-carried K loop.
static __device__ __forceinline__ void gemm4_core(
    const h16* __restrict__ A, long lda, long r0,
    const h16* __restrict__ Bt, long ldbt, long n0,
    long Kpad, int lane, v8f (&acc)[4])
{
  const h16* pa  = A  + (r0 + (lane & 15)) * lda  + ((lane >> 4) ? 16 : 0);
  const h16* pb0 = Bt + (n0 + (lane & 15)) * ldbt + ((lane >> 4) ? 16 : 0);
  const h16* pb1 = pb0 + 16 * ldbt;
  const h16* pb2 = pb0 + 32 * ldbt;
  const h16* pb3 = pb0 + 48 * ldbt;
  v16h a  = ldA(pa);
  v16h b0 = *(const v16h*)pb0;
  v16h b1 = *(const v16h*)pb1;
  v16h b2 = *(const v16h*)pb2;
  v16h b3 = *(const v16h*)pb3;
  for (long i = 1, steps = Kpad >> 5; i < steps; ++i){
    pa += 32; pb0 += 32; pb1 += 32; pb2 += 32; pb3 += 32;
    v16h an = ldA(pa);
    v16h c0 = *(const v16h*)pb0;
    v16h c1 = *(const v16h*)pb1;
    v16h c2 = *(const v16h*)pb2;
    v16h c3 = *(const v16h*)pb3;
    acc[0] = WMMA16(a, b0, acc[0]);
    acc[1] = WMMA16(a, b1, acc[1]);
    acc[2] = WMMA16(a, b2, acc[2]);
    acc[3] = WMMA16(a, b3, acc[3]);
    a = an; b0 = c0; b1 = c1; b2 = c2; b3 = c3;
  }
  acc[0] = WMMA16(a, b0, acc[0]);
  acc[1] = WMMA16(a, b1, acc[1]);
  acc[2] = WMMA16(a, b2, acc[2]);
  acc[3] = WMMA16(a, b3, acc[3]);
}

// 32x64 output per wave (2 row-tiles x 4 col-tiles), double-buffered,
// pointer-carried. 12 load-pairs feed 8 WMMAs per K-step.
static __device__ __forceinline__ void gemm8_core(
    const h16* __restrict__ A, long lda, long r0,
    const h16* __restrict__ Bt, long ldbt, long n0,
    long Kpad, int lane, v8f (&acc)[8])
{
  const h16* pa0 = A  + (r0 + (lane & 15)) * lda  + ((lane >> 4) ? 16 : 0);
  const h16* pa1 = pa0 + 16 * lda;
  const h16* pb0 = Bt + (n0 + (lane & 15)) * ldbt + ((lane >> 4) ? 16 : 0);
  const h16* pb1 = pb0 + 16 * ldbt;
  const h16* pb2 = pb0 + 32 * ldbt;
  const h16* pb3 = pb0 + 48 * ldbt;
  v16h a0 = ldA(pa0);
  v16h a1 = ldA(pa1);
  v16h b0 = *(const v16h*)pb0;
  v16h b1 = *(const v16h*)pb1;
  v16h b2 = *(const v16h*)pb2;
  v16h b3 = *(const v16h*)pb3;
  for (long i = 1, steps = Kpad >> 5; i < steps; ++i){
    pa0 += 32; pa1 += 32; pb0 += 32; pb1 += 32; pb2 += 32; pb3 += 32;
    v16h A0 = ldA(pa0);
    v16h A1 = ldA(pa1);
    v16h B0 = *(const v16h*)pb0;
    v16h B1 = *(const v16h*)pb1;
    v16h B2 = *(const v16h*)pb2;
    v16h B3 = *(const v16h*)pb3;
    acc[0] = WMMA16(a0, b0, acc[0]);
    acc[1] = WMMA16(a0, b1, acc[1]);
    acc[2] = WMMA16(a0, b2, acc[2]);
    acc[3] = WMMA16(a0, b3, acc[3]);
    acc[4] = WMMA16(a1, b0, acc[4]);
    acc[5] = WMMA16(a1, b1, acc[5]);
    acc[6] = WMMA16(a1, b2, acc[6]);
    acc[7] = WMMA16(a1, b3, acc[7]);
    a0 = A0; a1 = A1; b0 = B0; b1 = B1; b2 = B2; b3 = B3;
  }
  acc[0] = WMMA16(a0, b0, acc[0]);
  acc[1] = WMMA16(a0, b1, acc[1]);
  acc[2] = WMMA16(a0, b2, acc[2]);
  acc[3] = WMMA16(a0, b3, acc[3]);
  acc[4] = WMMA16(a1, b0, acc[4]);
  acc[5] = WMMA16(a1, b1, acc[5]);
  acc[6] = WMMA16(a1, b2, acc[6]);
  acc[7] = WMMA16(a1, b3, acc[7]);
}

// ---------------- weight prep: f32 [K x N] -> f16 transposed [Npad x Kpad] -----
__global__ void __launch_bounds__(256) k_prepB(
    const float* __restrict__ B, int K, int N,
    h16* __restrict__ Bt, long Kpad, long Npad)
{
  long id = (long)blockIdx.x * blockDim.x + threadIdx.x;
  if (id >= Npad * Kpad) return;
  long n = id / Kpad, k = id % Kpad;
  float v = (n < N && k < K) ? B[k * (long)N + n] : 0.0f;
  Bt[id] = (h16)v;
}

// conv weights [125][256][Ksz] -> Bt[128][Ksz*256] with row j = kk*256+ic
__global__ void __launch_bounds__(256) k_prepConv(
    const float* __restrict__ w, int Ksz, h16* __restrict__ Bt)
{
  long KD = (long)Ksz * 256;
  long id = (long)blockIdx.x * blockDim.x + threadIdx.x;
  if (id >= 128 * KD) return;
  long oc = id / KD, j = id % KD;
  long kk = j >> 8, ic = j & 255;
  float v = (oc < 125) ? w[oc * 256 * Ksz + ic * Ksz + kk] : 0.0f;
  Bt[id] = (h16)v;
}

// ---------------- generic WMMA GEMM (32x64 per wave; M multiple of 32) --------
// Epilogues: f32 out (+bias)(+relu), or f16 out with optional residual add.
__global__ void __launch_bounds__(256) k_gemm8(
    const h16* __restrict__ A, long lda,
    const h16* __restrict__ Bt, long ldbt,
    int M, int N, long Kpad,
    const float* __restrict__ bias, int relu,
    float* __restrict__ Cf, long ldc,
    const h16* __restrict__ resid, h16* __restrict__ Ch, long ldch)
{
  int wid = blockIdx.x * (blockDim.x >> 5) + ((int)threadIdx.x >> 5);
  int tilesM  = M >> 5;
  int groupsN = (N + 63) >> 6;
  if (wid >= tilesM * groupsN) return;
  long r0 = (long)(wid / groupsN) * 32;
  long n0 = (long)(wid % groupsN) * 64;
  int lane = threadIdx.x & 31;
  v8f acc[8] = {};
  gemm8_core(A, lda, r0, Bt, ldbt, n0, Kpad, lane, acc);
  int rb = (lane >> 4) ? 8 : 0;
#pragma unroll
  for (int rt = 0; rt < 2; ++rt){
#pragma unroll
    for (int tt = 0; tt < 4; ++tt){
      long col = n0 + tt * 16 + (lane & 15);
      if (col >= N) continue;
      float bb = bias ? bias[col] : 0.0f;
#pragma unroll
      for (int v = 0; v < 8; ++v){
        long row = r0 + rt * 16 + rb + v;
        float val = acc[rt * 4 + tt][v] + bb;
        if (relu) val = fmaxf(val, 0.0f);
        if (Ch){
          if (resid) val += (float)resid[row * ldch + col];
          Ch[row * ldch + col] = (h16)val;
        } else {
          Cf[row * ldc + col] = val;
        }
      }
    }
  }
}

// ---------------- embedding gather (f32 -> f16) ----------------
__global__ void __launch_bounds__(256) k_embed(
    const float* __restrict__ emb, const int* __restrict__ text2,
    h16* __restrict__ t0, long rows)
{
  long id = (long)blockIdx.x * blockDim.x + threadIdx.x;   // over rows*64
  if (id >= rows * 64) return;
  long r = id >> 6; int c4 = (int)(id & 63);
  int tok = text2[r];
  float4 f = ((const float4*)(emb + (long)tok * 256))[c4];
  h16* q = t0 + r * 256 + (long)c4 * 4;
  q[0] = (h16)f.x; q[1] = (h16)f.y; q[2] = (h16)f.z; q[3] = (h16)f.w;
}

__global__ void __launch_bounds__(256) k_f32_to_h16(
    const float* __restrict__ x, h16* __restrict__ y, long n){
  long id = (long)blockIdx.x * blockDim.x + threadIdx.x;
  if (id < n) y[id] = (h16)x[id];
}

// ---------------- fused per-node multi-head attention ----------------
#define ATT_NODES 2
__global__ void __launch_bounds__(64) k_attn(
    const h16* __restrict__ t0,
    const h16* __restrict__ WqT, const float* __restrict__ bq,
    const h16* __restrict__ WkT, const float* __restrict__ bk,
    const h16* __restrict__ WvT, const float* __restrict__ bv,
    h16* __restrict__ ctx, int Nn)
{
  __shared__ h16 qkv[ATT_NODES][3][16][256];
  int wave = (int)threadIdx.x >> 5;
  int lane = threadIdx.x & 31;
  long node = (long)blockIdx.x * ATT_NODES + wave;
  if (node >= Nn) return;
  const h16* X = t0 + node * 16 * 256;
  const h16*  WT_[3] = {WqT, WkT, WvT};
  const float* bs_[3] = {bq, bk, bv};

  for (int m = 0; m < 3; ++m){
    for (int g = 0; g < 4; ++g){
      v8f acc[4] = {};
      gemm4_core(X, 256, 0, WT_[m], 256, (long)g * 64, 256, lane, acc);
      int rb = (lane >> 4) ? 8 : 0;
#pragma unroll
      for (int tt = 0; tt < 4; ++tt){
        int col = g * 64 + tt * 16 + (lane & 15);
        float bb = bs_[m][col];
#pragma unroll
        for (int v = 0; v < 8; ++v)
          qkv[wave][m][rb + v][col] = (h16)(acc[tt][v] + bb);
      }
    }
  }
  // attention (per-wave private LDS region; DS ops in-order per wave)
  int qrow = lane & 15, half = lane >> 4;
  for (int h = 0; h < 16; ++h){
    int cb = h * 16;
    float s[8];
    float mx = -1e30f;
#pragma unroll
    for (int j = 0; j < 8; ++j){
      int key = half * 8 + j;
      float d = 0.0f;
#pragma unroll
      for (int dd = 0; dd < 16; ++dd)
        d += (float)qkv[wave][0][qrow][cb + dd] * (float)qkv[wave][1][key][cb + dd];
      s[j] = d * 0.25f;                         // 1/sqrt(dh=16)
      mx = fmaxf(mx, s[j]);
    }
    mx = fmaxf(mx, __shfl_xor(mx, 16, 32));
    float sum = 0.0f;
#pragma unroll
    for (int j = 0; j < 8; ++j){ s[j] = expf(s[j] - mx); sum += s[j]; }
    sum += __shfl_xor(sum, 16, 32);
    float inv = 1.0f / sum;
    float o[16];
#pragma unroll
    for (int dd = 0; dd < 16; ++dd){
      float a = 0.0f;
#pragma unroll
      for (int j = 0; j < 8; ++j)
        a += s[j] * (float)qkv[wave][2][half * 8 + j][cb + dd];
      o[dd] = a * inv;
    }
#pragma unroll
    for (int dd = 0; dd < 16; ++dd) o[dd] += __shfl_xor(o[dd], 16, 32);
    if (half == 0){
      h16* dst = ctx + node * 16 * 256 + (long)qrow * 256 + cb;
#pragma unroll
      for (int dd = 0; dd < 16; ++dd) dst[dd] = (h16)o[dd];
    }
  }
}

// ---------------- layernorm per row (wave per row, 256 cols) ----------------
__global__ void __launch_bounds__(256) k_layernorm(
    const h16* __restrict__ y, const float* __restrict__ g,
    const float* __restrict__ be, h16* __restrict__ t1, long rows)
{
  long row = (long)blockIdx.x * (blockDim.x >> 5) + ((int)threadIdx.x >> 5);
  if (row >= rows) return;
  int lane = threadIdx.x & 31;
  const h16* p = y + row * 256;
  float vals[8]; float s = 0.0f, s2 = 0.0f;
#pragma unroll
  for (int i = 0; i < 8; ++i){
    float v = (float)p[lane + 32 * i];
    vals[i] = v; s += v; s2 += v * v;
  }
#pragma unroll
  for (int off = 16; off >= 1; off >>= 1){
    s  += __shfl_xor(s,  off, 32);
    s2 += __shfl_xor(s2, off, 32);
  }
  float mu  = s * (1.0f / 256.0f);
  float var = s2 * (1.0f / 256.0f) - mu * mu;
  float inv = rsqrtf(var + 1e-5f);
  h16* q = t1 + row * 256;
#pragma unroll
  for (int i = 0; i < 8; ++i){
    int d = lane + 32 * i;
    q[d] = (h16)((vals[i] - mu) * inv * g[d] + be[d]);
  }
}

__global__ void __launch_bounds__(256) k_tavg(
    const h16* __restrict__ t1, float* __restrict__ tavg, int Nn){
  long id = (long)blockIdx.x * blockDim.x + threadIdx.x;
  if (id >= (long)Nn * 256) return;
  long n = id >> 8; int d = (int)(id & 255);
  float s = 0.0f;
  for (int q = 0; q < 16; ++q) s += (float)t1[(n * 16 + q) * 256 + d];
  tavg[id] = s * (1.0f / 16.0f);
}

// ---------------- conv1d branch as WMMA GEMM + relu + maxpool ----------------
__global__ void __launch_bounds__(256) k_conv(
    const h16* __restrict__ t1,
    const h16* __restrict__ bt0, const h16* __restrict__ bt1,
    const h16* __restrict__ bt2, const h16* __restrict__ bt3,
    const float* __restrict__ cb0, const float* __restrict__ cb1,
    const float* __restrict__ cb2, const float* __restrict__ cb3,
    float* __restrict__ tfeat, int Nn)
{
  int wid = blockIdx.x * (blockDim.x >> 5) + ((int)threadIdx.x >> 5);
  if (wid >= Nn * 8) return;
  int grp  = wid & 1;            // 64-col group (oc 0..63 / 64..127)
  int kidx = (wid >> 1) & 3;
  long node = (long)(wid >> 3);
  const h16* Bt_[4] = {bt0, bt1, bt2, bt3};
  const float* Bb_[4] = {cb0, cb1, cb2, cb3};
  int Ksz = 5 + kidx;
  int P   = 17 - Ksz;            // valid output positions
  long KD = (long)Ksz * 256;
  const h16* A = t1 + node * 4096;     // rows = positions p (padded to 16)
  int lane = threadIdx.x & 31;
  v8f acc[4] = {};
  gemm4_core(A, 256, 0, Bt_[kidx], KD, (long)grp * 64, KD, lane, acc);
  int rb = (lane >> 4) ? 8 : 0;
#pragma unroll
  for (int tt = 0; tt < 4; ++tt){
    long col = (long)grp * 64 + tt * 16 + (lane & 15);
    bool cok = col < 125;
    float bb = cok ? Bb_[kidx][col] : 0.0f;
    float mx = -1e30f;
#pragma unroll
    for (int v = 0; v < 8; ++v){
      int p = rb + v;
      if (p < P) mx = fmaxf(mx, fmaxf(acc[tt][v] + bb, 0.0f));
    }
    mx = fmaxf(mx, __shfl_xor(mx, 16, 32));
    if (rb == 0 && cok)
      tfeat[node * 500 + (long)kidx * 125 + col] = mx;
  }
}

// ---------------- graph conv helpers ----------------
static __device__ __forceinline__ unsigned enc_f(float v){
  unsigned b = __float_as_uint(v);
  return b ^ ((b >> 31) ? 0xFFFFFFFFu : 0x80000000u);
}
static __device__ __forceinline__ float dec_f(unsigned k){
  unsigned b = k ^ ((k >> 31) ? 0x80000000u : 0xFFFFFFFFu);
  return __uint_as_float(b);
}

__global__ void __launch_bounds__(256) k_alpha(
    const float* __restrict__ h, const float* __restrict__ as_,
    const float* __restrict__ ad_, float* __restrict__ als,
    float* __restrict__ ald, int Nn)
{
  int n = blockIdx.x * blockDim.x + threadIdx.x;
  if (n >= Nn) return;
  const float* p = h + (long)n * 128;
  float s = 0.0f, d = 0.0f;
  for (int i = 0; i < 128; ++i){ float v = p[i]; s += v * as_[i]; d += v * ad_[i]; }
  als[n] = s; ald[n] = d;
}

__global__ void __launch_bounds__(256) k_init_gat(
    unsigned* __restrict__ menc, float* __restrict__ den, int Nn){
  int n = blockIdx.x * blockDim.x + threadIdx.x;
  if (n < Nn){ menc[n] = 0x007FFFFFu; den[n] = 0.0f; }   // enc(-inf)
}

__global__ void __launch_bounds__(256) k_edge_max(
    const int* __restrict__ ei, int E, int Nt,
    const float* __restrict__ als, const float* __restrict__ ald,
    unsigned* __restrict__ menc)
{
  int e = blockIdx.x * blockDim.x + threadIdx.x;
  if (e >= Nt) return;
  int s = (e < E) ? ei[e] : (e - E);
  int d = (e < E) ? ei[E + e] : (e - E);
  float v = als[s] + ald[d];
  v = (v > 0.0f) ? v : 0.2f * v;
  atomicMax(&menc[d], enc_f(v));
}

__global__ void __launch_bounds__(256) k_edge_exp(
    const int* __restrict__ ei, int E, int Nt,
    const float* __restrict__ als, const float* __restrict__ ald,
    const unsigned* __restrict__ menc,
    float* __restrict__ exb, float* __restrict__ den)
{
  int e = blockIdx.x * blockDim.x + threadIdx.x;
  if (e >= Nt) return;
  int s = (e < E) ? ei[e] : (e - E);
  int d = (e < E) ? ei[E + e] : (e - E);
  float v = als[s] + ald[d];
  v = (v > 0.0f) ? v : 0.2f * v;
  float ex = expf(v - dec_f(menc[d]));
  exb[e] = ex;
  atomicAdd(&den[d], ex);
}

// wave per edge, 4 features per lane (F=128)
__global__ void __launch_bounds__(256) k_gat_scatter(
    const int* __restrict__ ei, int E, int Nt,
    const float* __restrict__ exb, const float* __restrict__ den,
    const float* __restrict__ h, float* __restrict__ out)
{
  int e = blockIdx.x * (blockDim.x >> 5) + ((int)threadIdx.x >> 5);
  if (e >= Nt) return;
  int s = (e < E) ? ei[e] : (e - E);
  int d = (e < E) ? ei[E + e] : (e - E);
  float coef = exb[e] / den[d];
  int lane = threadIdx.x & 31;
#pragma unroll
  for (int i = 0; i < 4; ++i){
    int f = lane * 4 + i;
    atomicAdd(&out[(long)d * 128 + f], coef * h[(long)s * 128 + f]);
  }
}

__global__ void __launch_bounds__(256) k_deg(
    const int* __restrict__ ei, int E, int Nt, float* __restrict__ deg){
  int e = blockIdx.x * blockDim.x + threadIdx.x;
  if (e >= Nt) return;
  int d = (e < E) ? ei[E + e] : (e - E);
  atomicAdd(&deg[d], 1.0f);
}

__global__ void __launch_bounds__(256) k_dinv(
    const float* __restrict__ deg, float* __restrict__ dinv, int Nn){
  int n = blockIdx.x * blockDim.x + threadIdx.x;
  if (n < Nn) dinv[n] = rsqrtf(fmaxf(deg[n], 1.0f));
}

__global__ void __launch_bounds__(256) k_gcn_scatter(
    const int* __restrict__ ei, int E, int Nt,
    const float* __restrict__ dinv,
    const float* __restrict__ h, float* __restrict__ out)
{
  int e = blockIdx.x * (blockDim.x >> 5) + ((int)threadIdx.x >> 5);
  if (e >= Nt) return;
  int s = (e < E) ? ei[e] : (e - E);
  int d = (e < E) ? ei[E + e] : (e - E);
  float coef = dinv[s] * dinv[d];
  int lane = threadIdx.x & 31;
#pragma unroll
  for (int i = 0; i < 4; ++i){
    int f = lane * 4 + i;
    atomicAdd(&out[(long)d * 128 + f], coef * h[(long)s * 128 + f]);
  }
}

// ---------------- concat / final-feature build ----------------
__global__ void __launch_bounds__(256) k_cat(
    const float* __restrict__ sg, const float* __restrict__ gb1,
    const float* __restrict__ ss, const float* __restrict__ cb1,
    const float* __restrict__ tavg, h16* __restrict__ cat, int Nn)
{
  long id = (long)blockIdx.x * blockDim.x + threadIdx.x;
  if (id >= (long)Nn * 512) return;
  long n = id >> 9; int c = (int)(id & 511);
  float v;
  if (c < 128)      v = sg[n * 128 + c] + gb1[c];
  else if (c < 256) v = ss[n * 128 + (c - 128)] + cb1[c - 128];
  else              v = tavg[n * 256 + (c - 256)];
  cat[id] = (h16)v;
}

__global__ void __launch_bounds__(256) k_final(
    const float* __restrict__ sg2, const float* __restrict__ gb2,
    const float* __restrict__ tfeat,
    const int* __restrict__ root, const int* __restrict__ rootpost,
    h16* __restrict__ fin)
{
  long id = (long)blockIdx.x * blockDim.x + threadIdx.x;
  if (id >= 1024L * 1152) return;
  long b = id / 1152; int c = (int)(id % 1152);
  float v = 0.0f;
  if (c < 128)       v = sg2[(long)root[b] * 128 + c] + gb2[c];
  else if (c < 628)  v = tfeat[(long)root[b] * 500 + (c - 128)];
  else if (c < 1128) v = tfeat[(long)rootpost[b] * 500 + (c - 628)];
  fin[id] = (h16)v;
}

__global__ void __launch_bounds__(256) k_fc2(
    const float* __restrict__ a, const float* __restrict__ W,
    const float* __restrict__ b, float* __restrict__ out)
{
  int id = blockIdx.x * blockDim.x + threadIdx.x;
  if (id >= 1024 * 3) return;
  int r = id / 3, c = id % 3;
  float s = b[c];
  for (int k = 0; k < 300; ++k) s += a[r * 300 + k] * W[k * 3 + c];
  out[id] = s;
}

// ================================================================
extern "C" void kernel_launch(void* const* d_in, const int* in_sizes, int n_in,
                              void* d_out, int out_size, void* d_ws, size_t ws_size,
                              hipStream_t stream)
{
  (void)in_sizes; (void)n_in; (void)out_size; (void)ws_size;
  const int  Nn = 32768, E = 131072, Nt = E + Nn;
  const long NS = (long)Nn * 16;

  const float* x2    = (const float*)d_in[0];
  const float* emb   = (const float*)d_in[1];
  const float* Wq    = (const float*)d_in[2];
  const float* Wk    = (const float*)d_in[3];
  const float* Wv    = (const float*)d_in[4];
  const float* Wo    = (const float*)d_in[5];
  const float* bq    = (const float*)d_in[6];
  const float* bk    = (const float*)d_in[7];
  const float* bv    = (const float*)d_in[8];
  const float* bo    = (const float*)d_in[9];
  const float* ln_g  = (const float*)d_in[10];
  const float* ln_b  = (const float*)d_in[11];
  const float* gat1_W  = (const float*)d_in[12];
  const float* gat1_as = (const float*)d_in[13];
  const float* gat1_ad = (const float*)d_in[14];
  const float* gat1_b  = (const float*)d_in[15];
  const float* gcn1_W  = (const float*)d_in[16];
  const float* gcn1_b  = (const float*)d_in[17];
  const float* gat2_W  = (const float*)d_in[18];
  const float* gat2_as = (const float*)d_in[19];
  const float* gat2_ad = (const float*)d_in[20];
  const float* gat2_b  = (const float*)d_in[21];
  const float* cw0 = (const float*)d_in[24]; const float* cbb0 = (const float*)d_in[25];
  const float* cw1 = (const float*)d_in[26]; const float* cbb1 = (const float*)d_in[27];
  const float* cw2 = (const float*)d_in[28]; const float* cbb2 = (const float*)d_in[29];
  const float* cw3 = (const float*)d_in[30]; const float* cbb3 = (const float*)d_in[31];
  const float* fc1_W = (const float*)d_in[32]; const float* fc1_b = (const float*)d_in[33];
  const float* fc2_W = (const float*)d_in[34]; const float* fc2_b = (const float*)d_in[35];
  const int* text2    = (const int*)d_in[36];
  const int* ei       = (const int*)d_in[37];
  const int* root     = (const int*)d_in[38];
  const int* rootpost = (const int*)d_in[39];
  float* out = (float*)d_out;

  // -------- workspace carve-out --------
  char* base = (char*)d_ws;
  size_t off = 0;
  auto alloc = [&](size_t bytes)->char*{
    char* p = base + off; off = (off + bytes + 255) & ~(size_t)255; return p; };
  h16*   t0   = (h16*)  alloc(NS * 256 * 2 + 4096);  // t0, later t1 (+conv row overrun pad)
  h16*   ctx  = (h16*)  alloc(NS * 256 * 2);
  h16*   ybuf = (h16*)  alloc(NS * 256 * 2);
  h16*   x2h  = (h16*)  alloc((size_t)Nn * 128 * 2);
  float* h1   = (float*)alloc((size_t)Nn * 128 * 4);
  float* h2   = (float*)alloc((size_t)Nn * 128 * 4);
  float* h3   = (float*)alloc((size_t)Nn * 128 * 4);
  float* sg   = (float*)alloc((size_t)Nn * 128 * 4);
  float* ss   = (float*)alloc((size_t)Nn * 128 * 4);
  float* sg2  = (float*)alloc((size_t)Nn * 128 * 4);
  float* tavg = (float*)alloc((size_t)Nn * 256 * 4);
  h16*   cat  = (h16*)  alloc((size_t)Nn * 512 * 2);
  float* tfeat= (float*)alloc((size_t)Nn * 500 * 4);
  float* als  = (float*)alloc((size_t)Nn * 4);
  float* ald  = (float*)alloc((size_t)Nn * 4);
  unsigned* menc = (unsigned*)alloc((size_t)Nn * 4);
  float* den  = (float*)alloc((size_t)Nn * 4);
  float* deg  = (float*)alloc((size_t)Nn * 4);
  float* dinv = (float*)alloc((size_t)Nn * 4);
  float* exb  = (float*)alloc((size_t)Nt * 4);
  h16*   finb = (h16*)  alloc(1024L * 1152 * 2);
  float* fc1o = (float*)alloc(1024L * 300 * 4);
  // transposed/padded f16 weights
  h16* WqT  = (h16*)alloc(256L * 256 * 2);
  h16* WkT  = (h16*)alloc(256L * 256 * 2);
  h16* WvT  = (h16*)alloc(256L * 256 * 2);
  h16* WoT  = (h16*)alloc(256L * 256 * 2);
  h16* g1T  = (h16*)alloc(128L * 128 * 2);
  h16* c1T  = (h16*)alloc(128L * 128 * 2);
  h16* g2T  = (h16*)alloc(128L * 512 * 2);
  h16* fc1T = (h16*)alloc(320L * 1152 * 2);
  h16* ct0  = (h16*)alloc(128L * 1280 * 2);
  h16* ct1  = (h16*)alloc(128L * 1536 * 2);
  h16* ct2  = (h16*)alloc(128L * 1792 * 2);
  h16* ct3  = (h16*)alloc(128L * 2048 * 2);
  h16* t1   = t0;   // reuse after residual consumed

  hipMemsetAsync(sg,  0, (size_t)Nn * 128 * 4, stream);
  hipMemsetAsync(ss,  0, (size_t)Nn * 128 * 4, stream);
  hipMemsetAsync(sg2, 0, (size_t)Nn * 128 * 4, stream);
  hipMemsetAsync(deg, 0, (size_t)Nn * 4, stream);

  // ---- weight prep ----
  auto prep = [&](const float* B, int K, int N, h16* Bt, long Kpad, long Npad){
    long tot = Npad * Kpad;
    k_prepB<<<(int)((tot + 255) / 256), 256, 0, stream>>>(B, K, N, Bt, Kpad, Npad);
  };
  prep(Wq, 256, 256, WqT, 256, 256);
  prep(Wk, 256, 256, WkT, 256, 256);
  prep(Wv, 256, 256, WvT, 256, 256);
  prep(Wo, 256, 256, WoT, 256, 256);
  prep(gat1_W, 128, 128, g1T, 128, 128);
  prep(gcn1_W, 128, 128, c1T, 128, 128);
  prep(gat2_W, 512, 128, g2T, 512, 128);
  prep(fc1_W, 1128, 300, fc1T, 1152, 320);
  k_prepConv<<<(int)((128L * 1280 + 255) / 256), 256, 0, stream>>>(cw0, 5, ct0);
  k_prepConv<<<(int)((128L * 1536 + 255) / 256), 256, 0, stream>>>(cw1, 6, ct1);
  k_prepConv<<<(int)((128L * 1792 + 255) / 256), 256, 0, stream>>>(cw2, 7, ct2);
  k_prepConv<<<(int)((128L * 2048 + 255) / 256), 256, 0, stream>>>(cw3, 8, ct3);

  auto gemm = [&](const h16* A, long lda, const h16* Bt, long ldbt,
                  int M, int N, long Kpad, const float* bias, int relu,
                  float* Cf, long ldc, const h16* resid, h16* Ch, long ldch){
    int waves  = (M >> 5) * ((N + 63) >> 6);
    int blocks = (waves + 7) / 8;
    k_gemm8<<<blocks, 256, 0, stream>>>(A, lda, Bt, ldbt, M, N, Kpad,
                                        bias, relu, Cf, ldc, resid, Ch, ldch);
  };

  // ---- input conversions / gathers ----
  k_f32_to_h16<<<(int)(((long)Nn * 128 + 255) / 256), 256, 0, stream>>>(x2, x2h, (long)Nn * 128);
  k_embed<<<(int)((NS * 64 + 255) / 256), 256, 0, stream>>>(emb, text2, t0, NS);

  // ---- GAT1 / GCN1 linear parts ----
  gemm(x2h, 128, g1T, 128, Nn, 128, 128, nullptr, 0, h1, 128, nullptr, nullptr, 0);
  gemm(x2h, 128, c1T, 128, Nn, 128, 128, nullptr, 0, h2, 128, nullptr, nullptr, 0);
  k_alpha<<<(Nn + 255) / 256, 256, 0, stream>>>(h1, gat1_as, gat1_ad, als, ald, Nn);

  // ---- GAT1 edge softmax + scatter ----
  k_init_gat<<<(Nn + 255) / 256, 256, 0, stream>>>(menc, den, Nn);
  k_edge_max<<<(Nt + 255) / 256, 256, 0, stream>>>(ei, E, Nt, als, ald, menc);
  k_edge_exp<<<(Nt + 255) / 256, 256, 0, stream>>>(ei, E, Nt, als, ald, menc, exb, den);
  k_gat_scatter<<<(Nt + 7) / 8, 256, 0, stream>>>(ei, E, Nt, exb, den, h1, sg);

  // ---- GCN1 ----
  k_deg<<<(Nt + 255) / 256, 256, 0, stream>>>(ei, E, Nt, deg);
  k_dinv<<<(Nn + 255) / 256, 256, 0, stream>>>(deg, dinv, Nn);
  k_gcn_scatter<<<(Nt + 7) / 8, 256, 0, stream>>>(ei, E, Nt, dinv, h2, ss);

  // ---- transformer ----
  k_attn<<<(Nn + ATT_NODES - 1) / ATT_NODES, 32 * ATT_NODES, 0, stream>>>(
      t0, WqT, bq, WkT, bk, WvT, bv, ctx, Nn);
  gemm(ctx, 256, WoT, 256, (int)NS, 256, 256, bo, 0, nullptr, 0, t0, ybuf, 256);
  k_layernorm<<<(int)((NS + 7) / 8), 256, 0, stream>>>(ybuf, ln_g, ln_b, t1, NS);
  k_tavg<<<(int)(((long)Nn * 256 + 255) / 256), 256, 0, stream>>>(t1, tavg, Nn);

  // ---- concat + GAT2 ----
  k_cat<<<(int)(((long)Nn * 512 + 255) / 256), 256, 0, stream>>>(sg, gat1_b, ss, gcn1_b, tavg, cat, Nn);
  gemm(cat, 512, g2T, 512, Nn, 128, 512, nullptr, 0, h3, 128, nullptr, nullptr, 0);
  k_alpha<<<(Nn + 255) / 256, 256, 0, stream>>>(h3, gat2_as, gat2_ad, als, ald, Nn);
  k_init_gat<<<(Nn + 255) / 256, 256, 0, stream>>>(menc, den, Nn);
  k_edge_max<<<(Nt + 255) / 256, 256, 0, stream>>>(ei, E, Nt, als, ald, menc);
  k_edge_exp<<<(Nt + 255) / 256, 256, 0, stream>>>(ei, E, Nt, als, ald, menc, exb, den);
  k_gat_scatter<<<(Nt + 7) / 8, 256, 0, stream>>>(ei, E, Nt, exb, den, h3, sg2);
  // (gcn2 output is dead in the reference forward — skipped)

  // ---- conv branch ----
  k_conv<<<(Nn * 8 + 7) / 8, 256, 0, stream>>>(t1, ct0, ct1, ct2, ct3,
                                               cbb0, cbb1, cbb2, cbb3, tfeat, Nn);

  // ---- head ----
  k_final<<<(int)((1024L * 1152 + 255) / 256), 256, 0, stream>>>(sg2, gat2_b, tfeat, root, rootpost, finb);
  gemm(finb, 1152, fc1T, 1152, 1024, 300, 1152, fc1_b, 1, fc1o, 300, nullptr, nullptr, 0);
  k_fc2<<<(1024 * 3 + 255) / 256, 256, 0, stream>>>(fc1o, fc2_W, fc2_b, out);
}
